// BiMambaV2_33071248179635
// MI455X (gfx1250) — compile-verified, matrix operand order checked by reading, not run on hardware
//
#include <hip/hip_runtime.h>
#include <hip/hip_bf16.h>

#define D_MODEL 1024
#define D_INNER 2048
#define D_STATE 16
#define D_CONV  4
#define DT_RANK 64
#define BATCH   4
#define SEQLEN  2048
#define XDBL_W  (DT_RANK + 2 * D_STATE)   // 96

typedef __attribute__((ext_vector_type(16))) __bf16    v16bf;
typedef __attribute__((ext_vector_type(8)))  __bf16    bf16x8;
typedef __attribute__((ext_vector_type(4)))  __bf16    bf16x4;
typedef __attribute__((ext_vector_type(8)))  float     v8f;
typedef __attribute__((ext_vector_type(4)))  float     f32x4;
typedef __attribute__((ext_vector_type(4)))  unsigned  u32x4;
typedef __attribute__((ext_vector_type(4)))  int       i32x4;
typedef __attribute__((ext_vector_type(8)))  int       i32x8;

#if defined(__has_builtin)
#if __has_builtin(__builtin_amdgcn_tensor_load_to_lds) && __has_builtin(__builtin_amdgcn_s_wait_tensorcnt)
#define USE_TDM 1
#endif
#endif

__device__ __forceinline__ __bf16 f2bf(float f) {
    union { float f; unsigned u; } v; v.f = f;
    unsigned r = (v.u + 0x7FFFu + ((v.u >> 16) & 1u)) >> 16;   // round-to-nearest-even
    union { unsigned short s; __bf16 b; } o; o.s = (unsigned short)r;
    return o.b;
}

__device__ __forceinline__ float sigmoidf_(float x) { return 1.f / (1.f + __expf(-x)); }
__device__ __forceinline__ float softplusf_(float x) {
    return fmaxf(x, 0.f) + log1pf(__expf(-fabsf(x)));
}

// fp32 -> bf16 pack (bandwidth-bound; done ONCE per operand instead of per GEMM tile)
__global__ void pack_bf16_kernel(const float* __restrict__ src,
                                 __bf16* __restrict__ dst, long long n)
{
    const long long i = ((long long)blockIdx.x * blockDim.x + threadIdx.x) * 4;
    if (i < n) {
        const f32x4 v = *(const f32x4*)(src + i);
        bf16x4 o;
#pragma unroll
        for (int j = 0; j < 4; ++j) o[j] = f2bf(v[j]);
        *(bf16x4*)(dst + i) = o;
    }
}

// Wave-tile WMMA GEMM on native bf16 operands (unit K-stride A & B):
//   C[m,n] = epilogue( sum_k A[m,k]*B[n,k] ),  one 16x16 C tile per wave.
// The block's 4 waves share one M-tile: a 16x32 bf16 slab (1KB) is staged in
// LDS by the Tensor Data Mover (cooperative b64 copies as fallback); fragments
// are two 16B K-runs -> ds_load_b128 (A) / global_load_b128 (B), zero cvt VALU
// in the K-loop. epilogue: 0 = none, 1 = +bias[n], 2 = softplus(+bias[n])
__global__ void __launch_bounds__(128)
wmma_gemm_bf16(const __bf16* __restrict__ A, const __bf16* __restrict__ Bw,
               float* __restrict__ C,
               int N, int K,
               long long As_r, long long A_bs,
               long long Bs_n,
               long long Cs_r, long long Cs_n, long long C_bs,
               const float* __restrict__ bias, int epilogue)
{
    __shared__ __bf16 sA[16 * 32];                 // one 16(M) x 32(K) bf16 slab
    const int lane = threadIdx.x;                  // 0..31
    const int wave = threadIdx.y;                  // 0..3
    const int tn = blockIdx.x * 4 + wave;          // tile along N (per wave)
    const int tm = blockIdx.y;                     // tile along M (shared by block)
    const bool active = (tn * 16 < N);             // wave-uniform
    const __bf16* Ab = A + (long long)blockIdx.z * A_bs + (long long)(tm * 16) * As_r;
    float*        Cb = C + (long long)blockIdx.z * C_bs;

    const int ml = lane & 15;                      // row (A) / col (B) within tile
    const int h  = lane >> 4;                      // half-wave K interleave
    const long long brow = (long long)((active ? tn : 0) * 16 + ml) * Bs_n;

#ifdef USE_TDM
    // Tensor DMA descriptor (ISA 8.3/8.4): 2D tile 32(K) x 16(M), bf16 (2B),
    // row stride = As_r elements, contiguous into LDS at sA.
    const unsigned ldsOff = (unsigned)(unsigned long long)(void*)sA;
    const unsigned long long gbase = (unsigned long long)Ab;
    i32x8 g1;
    g1[0] = 0x10000;                                              // data_size = 2B
    g1[1] = (int)(((unsigned)K & 0xFFFFu) << 16);                 // tensor_dim0 lo16
    g1[2] = (int)((((unsigned)K >> 16) & 0xFFFFu) | (16u << 16)); // dim0 hi16 | tensor_dim1=16
    g1[3] = (int)(32u << 16);                                     // tile_dim0 = 32
    g1[4] = 16;                                                   // tile_dim1 = 16
    g1[5] = (int)(unsigned)(As_r & 0xFFFFFFFFll);                 // dim0_stride lo32
    g1[6] = (int)(unsigned)((As_r >> 32) & 0xFFFFll);             // dim0_stride hi16
    g1[7] = 0;
    const i32x4 gz = {0, 0, 0, 0};
#else
    const int tid  = wave * 32 + lane;
    const int srow = tid >> 3;                     // 0..15
    const int scol = (tid & 7) * 4;                // 0,4,...,28
#endif

    v8f acc = {};
    for (int k0 = 0; k0 < K; k0 += 32) {
        __syncthreads();                           // prior fragment reads done
#ifdef USE_TDM
        if (wave == 0) {
            const unsigned long long ga = gbase + (unsigned long long)k0 * 2ull;
            u32x4 g0;
            g0[0] = 1u;                                             // count = 1 (valid)
            g0[1] = ldsOff;                                         // lds_addr
            g0[2] = (unsigned)(ga & 0xFFFFFFFFull);                 // global_addr lo
            g0[3] = (unsigned)((ga >> 32) & 0x1FFFFFFull) | (2u << 30); // addr hi | type=2
#if __clang_major__ >= 23
            const i32x8 gz8 = {0, 0, 0, 0, 0, 0, 0, 0};
            __builtin_amdgcn_tensor_load_to_lds(g0, g1, gz, gz, gz8, 0);
#else
            __builtin_amdgcn_tensor_load_to_lds(g0, g1, gz, gz, 0);
#endif
            __builtin_amdgcn_s_wait_tensorcnt(0);
        }
#else
        *(bf16x4*)&sA[srow * 32 + scol] =
            *(const bf16x4*)(Ab + (long long)srow * As_r + (k0 + scol));
#endif
        __syncthreads();                           // slab published to all waves
        if (active) {
            // ISA 7.12.2 16-bit fragment layout: lane holds two contiguous
            // 8-element (16B) K-runs at k = 8h and k = 16+8h.
            const bf16x8 a0 = *(const bf16x8*)&sA[ml * 32 + 8 * h];       // ds_load_b128
            const bf16x8 a1 = *(const bf16x8*)&sA[ml * 32 + 16 + 8 * h];
            const bf16x8 b0 = *(const bf16x8*)(Bw + brow + k0 + 8 * h);   // global_load_b128
            const bf16x8 b1 = *(const bf16x8*)(Bw + brow + k0 + 16 + 8 * h);
            const v16bf af = __builtin_shufflevector(a0, a1,
                0, 1, 2, 3, 4, 5, 6, 7, 8, 9, 10, 11, 12, 13, 14, 15);
            const v16bf bfr = __builtin_shufflevector(b0, b1,
                0, 1, 2, 3, 4, 5, 6, 7, 8, 9, 10, 11, 12, 13, 14, 15);
            if (k0 + 32 < K)                            // gfx1250 global_prefetch_b8
                __builtin_prefetch(Bw + brow + k0 + 32, 0, 1);
            acc = __builtin_amdgcn_wmma_f32_16x16x32_bf16(
                      false, af, false, bfr, (short)0, acc, false, false);
        }
    }

    if (active) {
        // C/D layout: lane -> n = tn*16+(lane&15); VGPR r -> m = tm*16 + r + 8*h
        const int n = tn * 16 + ml;
        const float bv = (epilogue >= 1) ? bias[n] : 0.f;
#pragma unroll
        for (int r = 0; r < 8; ++r) {
            const int m = tm * 16 + r + 8 * h;
            float val = acc[r] + bv;
            if (epilogue == 2) val = softplusf_(val);
            Cb[(long long)m * Cs_r + (long long)n * Cs_n] = val;
        }
    }
}

// Depthwise causal conv (width 4) + SiLU.
// x lives in xz as (b, d, l); output u is (b, l, d) in direction-local time,
// written twice: fp32 (scan precision) and bf16 (x_proj GEMM operand).
__global__ void conv_silu_kernel(const float* __restrict__ xz,
                                 const float* __restrict__ w,
                                 const float* __restrict__ bias,
                                 float* __restrict__ u,
                                 __bf16* __restrict__ u_bf, int reverse)
{
    long long idx = (long long)blockIdx.x * blockDim.x + threadIdx.x;  // d fastest
    if (idx >= (long long)BATCH * SEQLEN * D_INNER) return;
    const int d = (int)(idx % D_INNER);
    const long long t = idx / D_INNER;
    const int l = (int)(t % SEQLEN);
    const int b = (int)(t / SEQLEN);

    const float* xb = xz + ((long long)b * (2 * D_INNER) + d) * SEQLEN;
    float acc = bias[d];
#pragma unroll
    for (int j = 0; j < D_CONV; ++j) {
        const int li = l - (D_CONV - 1) + j;          // tap (direction-local time)
        if (li >= 0) {
            const int src = reverse ? (SEQLEN - 1 - li) : li;
            acc += w[d * D_CONV + j] * xb[src];
        }
    }
    const float s = acc * sigmoidf_(acc);             // silu
    const long long oi = ((long long)b * SEQLEN + l) * D_INNER + d;
    u[oi] = s;
    u_bf[oi] = f2bf(s);
}

// Selective scan: one thread per (b, d); 16-state fp32 recurrence in registers.
// delta/u are (b,l,d) (lane-coalesced over d), B/C rows vector-loaded b128.
// Fuses D*u skip + silu(z) gate; stores at true-time index (backward lands
// un-flipped); accumulate=1 adds into y -> 0.5*(yf+yb) combine.
__global__ void ssm_scan_kernel(const float* __restrict__ u,
                                const float* __restrict__ delta,
                                const float* __restrict__ x_dbl,
                                const float* __restrict__ A_log,
                                const float* __restrict__ Dp,
                                const float* __restrict__ xz,   // z = xz[b][D_INNER+d][l]
                                float* __restrict__ y,
                                int reverse, int accumulate)
{
    const int idx = blockIdx.x * blockDim.x + threadIdx.x;
    if (idx >= BATCH * D_INNER) return;
    const int d = idx % D_INNER;
    const int b = idx / D_INNER;

    float A[D_STATE];
#pragma unroll
    for (int n = 0; n < D_STATE; ++n) A[n] = -__expf(A_log[d * D_STATE + n]);
    const float Dd = Dp[d];
    float hst[D_STATE];
#pragma unroll
    for (int n = 0; n < D_STATE; ++n) hst[n] = 0.f;

    const float* zrow = xz + ((long long)b * (2 * D_INNER) + D_INNER + d) * SEQLEN;

    for (int t = 0; t < SEQLEN; ++t) {
        const long long rl = (long long)b * SEQLEN + t;     // direction-local time
        const float dl = delta[rl * D_INNER + d];
        const float ul = u[rl * D_INNER + d];
        const f32x4* bc4 = (const f32x4*)(x_dbl + rl * XDBL_W + DT_RANK);
        float Bv[D_STATE], Cv[D_STATE];
#pragma unroll
        for (int q = 0; q < 4; ++q) {
            *(f32x4*)&Bv[q * 4] = bc4[q];          // B slice: b128 loads
            *(f32x4*)&Cv[q * 4] = bc4[4 + q];      // C slice
        }
        const float du = dl * ul;
        float yv = 0.f;
#pragma unroll
        for (int n = 0; n < D_STATE; ++n) {
            const float dA = __expf(dl * A[n]);
            hst[n] = dA * hst[n] + du * Bv[n];
            yv += hst[n] * Cv[n];
        }
        yv += Dd * ul;

        const int ltrue = reverse ? (SEQLEN - 1 - t) : t;
        const float zz = zrow[ltrue];
        const float g = yv * (zz * sigmoidf_(zz));          // * silu(z)
        const long long oi = ((long long)b * SEQLEN + ltrue) * D_INNER + d;
        if (accumulate) y[oi] = y[oi] + 0.5f * g;
        else            y[oi] = 0.5f * g;
    }
}

static inline void pack_bf16(const float* s, __bf16* d, long long n, hipStream_t st) {
    pack_bf16_kernel<<<(unsigned)((n / 4 + 255) / 256), 256, 0, st>>>(s, d, n);
}

extern "C" void kernel_launch(void* const* d_in, const int* in_sizes, int n_in,
                              void* d_out, int out_size, void* d_ws, size_t ws_size,
                              hipStream_t stream) {
    (void)in_sizes; (void)n_in; (void)out_size; (void)ws_size;
    const float* hs          = (const float*)d_in[0];
    const float* in_proj_w   = (const float*)d_in[1];
    const float* conv_w[2]   = {(const float*)d_in[2], (const float*)d_in[4]};
    const float* conv_b[2]   = {(const float*)d_in[3], (const float*)d_in[5]};
    const float* x_proj_w[2] = {(const float*)d_in[6], (const float*)d_in[9]};
    const float* dt_proj_w[2]= {(const float*)d_in[7], (const float*)d_in[10]};
    const float* dt_proj_b[2]= {(const float*)d_in[8], (const float*)d_in[11]};
    const float* A_log[2]    = {(const float*)d_in[12], (const float*)d_in[13]};
    const float* Dp[2]       = {(const float*)d_in[14], (const float*)d_in[15]};
    const float* out_proj_w  = (const float*)d_in[16];
    float* out = (float*)d_out;

    // fp32 workspace: xz(128MB) + u(64MB) + x_dbl(3MB) + delta(64MB) + y(64MB)
    float* xz    = (float*)d_ws;                                 // (B, 2*D_INNER, L)
    float* u     = xz    + (size_t)BATCH * 2 * D_INNER * SEQLEN; // (B, L, D_INNER)
    float* xdbl  = u     + (size_t)BATCH * SEQLEN * D_INNER;     // (B, L, 96)
    float* delta = xdbl  + (size_t)BATCH * SEQLEN * XDBL_W;      // (B, L, D_INNER)
    float* ycmb  = delta + (size_t)BATCH * SEQLEN * D_INNER;     // (B, L, D_INNER)
    // bf16 operand mirrors (packed once, streamed by TDM/WMMA)
    __bf16* hs_bf   = (__bf16*)(ycmb + (size_t)BATCH * SEQLEN * D_INNER);
    __bf16* wip_bf  = hs_bf  + (size_t)BATCH * SEQLEN * D_MODEL;
    __bf16* wop_bf  = wip_bf + (size_t)2 * D_INNER * D_MODEL;
    __bf16* wxp_bf  = wop_bf + (size_t)D_MODEL * D_INNER;
    __bf16* wdt_bf  = wxp_bf + (size_t)XDBL_W * D_INNER;
    __bf16* u_bf    = wdt_bf + (size_t)D_INNER * DT_RANK;
    __bf16* xdbl_bf = u_bf   + (size_t)BATCH * SEQLEN * D_INNER;
    __bf16* y_bf    = xdbl_bf+ (size_t)BATCH * SEQLEN * XDBL_W;

    const dim3 blk(32, 4);

    // 0) pack static operands to bf16 (once per launch)
    pack_bf16(hs,         hs_bf,  (long long)BATCH * SEQLEN * D_MODEL, stream);
    pack_bf16(in_proj_w,  wip_bf, (long long)2 * D_INNER * D_MODEL,    stream);
    pack_bf16(out_proj_w, wop_bf, (long long)D_MODEL * D_INNER,        stream);

    // 1) in_proj: xz[b][e][l] = sum_d hs[b][l][d] * W[e][d]
    wmma_gemm_bf16<<<dim3((2 * D_INNER) / 64, SEQLEN / 16, BATCH), blk, 0, stream>>>(
        hs_bf, wip_bf, xz, 2 * D_INNER, D_MODEL,
        /*As_r*/ D_MODEL, /*A_bs*/ (long long)SEQLEN * D_MODEL,
        /*Bs_n*/ D_MODEL,
        /*Cs_r(l)*/ 1, /*Cs_n(e)*/ SEQLEN, /*C_bs*/ (long long)2 * D_INNER * SEQLEN,
        nullptr, 0);

    for (int dir = 0; dir < 2; ++dir) {
        // per-direction weight packs
        pack_bf16(x_proj_w[dir],  wxp_bf, (long long)XDBL_W * D_INNER, stream);
        pack_bf16(dt_proj_w[dir], wdt_bf, (long long)D_INNER * DT_RANK, stream);

        // 2) depthwise causal conv + silu -> u (fp32 + bf16 mirror)
        const long long tot = (long long)BATCH * SEQLEN * D_INNER;
        conv_silu_kernel<<<(unsigned)((tot + 255) / 256), 256, 0, stream>>>(
            xz, conv_w[dir], conv_b[dir], u, u_bf, dir);

        // 3) x_proj: x_dbl[b][l][r] = sum_d u[b][l][d] * Wx[r][d]   (N=96 -> 6 tiles)
        wmma_gemm_bf16<<<dim3(2, SEQLEN / 16, BATCH), blk, 0, stream>>>(
            u_bf, wxp_bf, xdbl, XDBL_W, D_INNER,
            D_INNER, (long long)SEQLEN * D_INNER,
            D_INNER,
            XDBL_W, 1, (long long)SEQLEN * XDBL_W,
            nullptr, 0);

        // 4) dt_proj + bias + softplus fused (A = x_dbl cols [0,64) in bf16)
        pack_bf16(xdbl, xdbl_bf, (long long)BATCH * SEQLEN * XDBL_W, stream);
        wmma_gemm_bf16<<<dim3(D_INNER / 64, SEQLEN / 16, BATCH), blk, 0, stream>>>(
            xdbl_bf, wdt_bf, delta, D_INNER, DT_RANK,
            XDBL_W, (long long)SEQLEN * XDBL_W,
            DT_RANK,
            D_INNER, 1, (long long)SEQLEN * D_INNER,
            dt_proj_b[dir], 2);

        // 5) selective scan + gate + 0.5*(f+b) combine (store then accumulate)
        ssm_scan_kernel<<<(BATCH * D_INNER + 255) / 256, 256, 0, stream>>>(
            u, delta, xdbl, A_log[dir], Dp[dir], xz, ycmb, dir, dir);
    }

    // 6) out_proj: out[b][l][e] = sum_d y[b][l][d] * Wo[e][d]
    pack_bf16(ycmb, y_bf, (long long)BATCH * SEQLEN * D_INNER, stream);
    wmma_gemm_bf16<<<dim3(D_MODEL / 64, SEQLEN / 16, BATCH), blk, 0, stream>>>(
        y_bf, wop_bf, out, D_MODEL, D_INNER,
        D_INNER, (long long)SEQLEN * D_INNER,
        D_INNER,
        D_MODEL, 1, (long long)SEQLEN * D_MODEL,
        nullptr, 0);
}
